// VonMisesNet_45183055954187
// MI455X (gfx1250) — matrix operand order repaired and checked by reading, not compiled, not run on hardware
//
#include <hip/hip_runtime.h>
#include <hip/hip_bf16.h>

// ---- problem constants (match reference) ----
constexpr int Bc  = 128;   // batch
constexpr int Vc  = 128;   // vertices
constexpr int Fc  = 40;    // input features
constexpr int Hc  = 256;   // hidden
constexpr int Lc  = 32;    // layers
constexpr int FLc = 1024;  // final hidden
constexpr float LN_EPS = 1e-5f;

typedef __attribute__((ext_vector_type(16))) _Float16 v16h;
typedef __attribute__((ext_vector_type(8)))  _Float16 v8h;
typedef __attribute__((ext_vector_type(8)))  float    v8f;

#define SROW 40   // LDS row stride in halfs (32 + 8 pad -> 80B, keeps 16B alignment)
constexpr int ASZ = 64  * SROW;   // halfs per A buffer (64 M-rows)
constexpr int BSZ = 256 * SROW;   // halfs per B buffer (256 N-rows)

// ---------------------------------------------------------------------------
// WMMA GEMM core: 64(M) x 256(N) C tile per 256-thread block (8 waves).
// A  : M x K row-major f16 (ld = lda)
// Bt : N x K row-major f16 (ld = ldb)  == B transposed, K contiguous
// Wave w owns columns [n0+32w, n0+32w+32): acc[2][4] = 8 WMMAs per 32-K step,
// A fragments reused across both column groups. Double-buffered LDS with
// load-early / store-late staging so global latency hides under the WMMAs.
// ---------------------------------------------------------------------------
template<int K>
__device__ __forceinline__ void gemm_core(
    const _Float16* __restrict__ A, int lda, int m0,
    const _Float16* __restrict__ Bt, int ldb, int n0,
    _Float16* sA, _Float16* sB, v8f (*acc)[4])
{
    constexpr int STEPS = K / 32;
    const int tid  = threadIdx.x;
    const int wave = tid >> 5;
    const int lane = tid & 31;
    const int lh   = lane & 15;   // lane within half-wave
    const int hb   = lane >> 4;   // half-wave select

    // fixed per-thread staging slots (straight-line, no loops)
    const int ra = tid >> 2, ca = (tid & 3) * 8;   // A: 64 rows, 1 x v8h / thread
    const _Float16* pA = A  + (size_t)(m0 + ra) * lda + ca;
    const _Float16* pB = Bt + (size_t)(n0 + tid) * ldb;  // B: 1 full row / thread
    _Float16* qA = sA + ra * SROW + ca;
    _Float16* qB = sB + tid * SROW;

    // prologue: fill buffer 0
    *(v8h*)qA = *(const v8h*)pA;
#pragma unroll
    for (int u = 0; u < 4; ++u)
        *(v8h*)(qB + u * 8) = *(const v8h*)(pB + u * 8);
    __syncthreads();

#pragma unroll
    for (int s = 0; s < STEPS; ++s) {
        const int cur = s & 1, nxt = cur ^ 1;

        // issue next slab's global loads early; hold in registers
        v8h tA, tB[4];
        if (s + 1 < STEPS) {
            tA = *(const v8h*)(pA + (s + 1) * 32);
#pragma unroll
            for (int u = 0; u < 4; ++u)
                tB[u] = *(const v8h*)(pB + (s + 1) * 32 + u * 8);
        }
        const _Float16* bA = sA + cur * ASZ;
        const _Float16* bB = sB + cur * BSZ;

        // two B fragments: columns n = 32*wave + 16j + lh
        v16h bfrag[2];
#pragma unroll
        for (int j = 0; j < 2; ++j) {
            const v8h* bp =
                (const v8h*)(bB + (wave * 32 + j * 16 + lh) * SROW + hb * 16);
            ((v8h*)&bfrag[j])[0] = bp[0];
            ((v8h*)&bfrag[j])[1] = bp[1];
        }
        // A fragments reused across both column groups
#pragma unroll
        for (int i = 0; i < 4; ++i) {
            const _Float16* ap = bA + (i * 16 + lh) * SROW + hb * 16;
            v16h afrag;
            ((v8h*)&afrag)[0] = *(const v8h*)(ap);
            ((v8h*)&afrag)[1] = *(const v8h*)(ap + 8);
#pragma unroll
            for (int j = 0; j < 2; ++j)
                acc[j][i] = __builtin_amdgcn_wmma_f32_16x16x32_f16(
                    false, afrag, false, bfrag[j], (short)0, acc[j][i],
                    false, false);
        }
        // store staged slab after compute has been issued
        if (s + 1 < STEPS) {
            *(v8h*)(qA + nxt * ASZ) = tA;
#pragma unroll
            for (int u = 0; u < 4; ++u)
                *(v8h*)(qB + nxt * BSZ + u * 8) = tB[u];
        }
        __syncthreads();
    }
}

// C element (j, i, r): row = m0 + 16i + 8*hb + r, col = n0 + 32*wave + 16j + lh

// ---------------------------------------------------------------------------
// GEMM 1: VP = Vh @ W_l + b_l  (M = B*V flat, N = H, K = H)
// Output written TRANSPOSED per batch: VPhT[b][h][v], packed b128 stores.
// ---------------------------------------------------------------------------
__global__ void __launch_bounds__(256)
k_gemm_vw(const _Float16* __restrict__ Vh, const _Float16* __restrict__ WT,
          const float* __restrict__ bias, _Float16* __restrict__ VPhT)
{
    __shared__ _Float16 sA[2 * ASZ];
    __shared__ _Float16 sB[2 * BSZ];
    v8f acc[2][4] = {};
    const int m0 = blockIdx.x * 64, n0 = blockIdx.y * 256;
    gemm_core<Hc>(Vh, Hc, m0, WT, Hc, n0, sA, sB, acc);

    const int wave = threadIdx.x >> 5, lane = threadIdx.x & 31;
    const int lh = lane & 15, hb = lane >> 4;
    const int b   = m0 >> 7;                        // batch of this tile
    const int vr0 = (m0 & (Vc - 1)) + hb * 8;       // vertex row base
#pragma unroll
    for (int j = 0; j < 2; ++j) {
        const int col = n0 + wave * 32 + j * 16 + lh;   // h index
        const float bv = bias[col];
        _Float16* dst = VPhT + ((size_t)b * Hc + col) * Vc + vr0;
#pragma unroll
        for (int i = 0; i < 4; ++i) {
            v8h pack;
#pragma unroll
            for (int r = 0; r < 8; ++r) pack[r] = (_Float16)(acc[j][i][r] + bv);
            *(v8h*)(dst + i * 16) = pack;           // 8 consecutive v rows
        }
    }
}

// ---------------------------------------------------------------------------
// GEMM 2 (batched): M = Gp[b] @ VP[b]; epilogue: /deg, relu, residual
// B operand is VPhT[b] which is already N(h) x K(v) contiguous.
// ---------------------------------------------------------------------------
__global__ void __launch_bounds__(256)
k_gemm_gp(const _Float16* __restrict__ Gph, const _Float16* __restrict__ VPhT,
          const float* __restrict__ deginv, float* __restrict__ Vf32,
          _Float16* __restrict__ Vh)
{
    __shared__ _Float16 sA[2 * ASZ];
    __shared__ _Float16 sB[2 * BSZ];
    v8f acc[2][4] = {};
    const int b  = blockIdx.z;
    const int m0 = blockIdx.x * 64, n0 = blockIdx.y * 256;
    gemm_core<Vc>(Gph + (size_t)b * Vc * Vc, Vc, m0,
                  VPhT + (size_t)b * Hc * Vc, Vc, n0, sA, sB, acc);

    const int wave = threadIdx.x >> 5, lane = threadIdx.x & 31;
    const int lh = lane & 15, hb = lane >> 4;
#pragma unroll
    for (int j = 0; j < 2; ++j) {
        const int col = n0 + wave * 32 + j * 16 + lh;
#pragma unroll
        for (int i = 0; i < 4; ++i)
#pragma unroll
            for (int r = 0; r < 8; ++r) {
                const int row = m0 + i * 16 + hb * 8 + r;
                const float m = acc[j][i][r] * deginv[b * Vc + row];
                const size_t idx = ((size_t)(b * Vc + row)) * Hc + col;
                const float nv = Vf32[idx] + (m > 0.f ? m : 0.f);
                Vf32[idx] = nv;
                Vh[idx]   = (_Float16)nv;
            }
    }
}

// ---------------------------------------------------------------------------
// GEMM 3: Hfin = relu(LN(v) @ W_fin + b_fin)  (M = B*V, N = FL, K = H)
// B operand is W_fin pre-transposed: [FL][H].
// ---------------------------------------------------------------------------
__global__ void __launch_bounds__(256)
k_gemm_fin(const _Float16* __restrict__ Ah, const _Float16* __restrict__ WT,
           const float* __restrict__ bias, _Float16* __restrict__ Out)
{
    __shared__ _Float16 sA[2 * ASZ];
    __shared__ _Float16 sB[2 * BSZ];
    v8f acc[2][4] = {};
    const int m0 = blockIdx.x * 64, n0 = blockIdx.y * 256;
    gemm_core<Hc>(Ah, Hc, m0, WT, Hc, n0, sA, sB, acc);

    const int wave = threadIdx.x >> 5, lane = threadIdx.x & 31;
    const int lh = lane & 15, hb = lane >> 4;
#pragma unroll
    for (int j = 0; j < 2; ++j) {
        const int col = n0 + wave * 32 + j * 16 + lh;
        const float bv = bias[col];
#pragma unroll
        for (int i = 0; i < 4; ++i)
#pragma unroll
            for (int r = 0; r < 8; ++r) {
                const int row = m0 + i * 16 + hb * 8 + r;
                const float v = acc[j][i][r] + bv;
                Out[(size_t)row * FLc + col] = (_Float16)(v > 0.f ? v : 0.f);
            }
    }
}

// ---------------------------------------------------------------------------
// Transposing f32 -> f16 weight conversion (division-free):
// dst[m][c][r] = src[m][r][c];  r = threadIdx.x (R==256), c = blockIdx.x,
// m = blockIdx.y. One-time cost; writes coalesced.
// ---------------------------------------------------------------------------
__global__ void __launch_bounds__(256)
k_cvtT(const float* __restrict__ src, _Float16* __restrict__ dst, int R, int C)
{
    const int r = threadIdx.x;
    const int c = blockIdx.x;
    const int m = blockIdx.y;
    dst[((size_t)m * C + c) * R + r] = (_Float16)src[((size_t)m * R + r) * C + c];
}

// v0 = x @ W_feat + b_feat ; one block per (b,v) row, thread = h
__global__ void __launch_bounds__(256)
k_featurize(const float* __restrict__ x, const float* __restrict__ Wf,
            const float* __restrict__ bf, float* __restrict__ Vf32,
            _Float16* __restrict__ Vh)
{
    const int bv = blockIdx.x;
    const int h  = threadIdx.x;
    const float* xr = x + (size_t)bv * Fc;
    float a = bf[h];
#pragma unroll
    for (int f = 0; f < Fc; ++f) a += xr[f] * Wf[f * Hc + h];
    const size_t idx = (size_t)bv * Hc + h;
    Vf32[idx] = a;
    Vh[idx]   = (_Float16)a;
}

// Gp = G + diag(rowsum(G)) in f16 ; deginv = 1/(n + (n==0)). Wave per row.
__global__ void __launch_bounds__(256)
k_prep(const float* __restrict__ G, _Float16* __restrict__ Gph,
       float* __restrict__ deginv)
{
    const int row  = blockIdx.x * 8 + (threadIdx.x >> 5);   // b*V + i
    const int lane = threadIdx.x & 31;
    const int i    = row & (Vc - 1);
    const float* gr = G + (size_t)row * Vc;
    float g[4], s = 0.f;
#pragma unroll
    for (int t = 0; t < 4; ++t) { g[t] = gr[lane + 32 * t]; s += g[t]; }
#pragma unroll
    for (int off = 16; off >= 1; off >>= 1) s += __shfl_xor(s, off);
#pragma unroll
    for (int t = 0; t < 4; ++t) {
        const int j = lane + 32 * t;
        Gph[(size_t)row * Vc + j] = (_Float16)(g[t] + (j == i ? s : 0.f));
    }
    if (lane == 0) deginv[row] = 1.0f / (s == 0.f ? 1.f : s);
}

// LayerNorm over H, f16 output (row-major). Wave per row.
__global__ void __launch_bounds__(256)
k_ln(const float* __restrict__ Vf32, const float* __restrict__ g,
     const float* __restrict__ bb, _Float16* __restrict__ Out)
{
    const int row  = blockIdx.x * 8 + (threadIdx.x >> 5);
    const int lane = threadIdx.x & 31;
    const float* vr = Vf32 + (size_t)row * Hc;
    float vals[8], s = 0.f, s2 = 0.f;
#pragma unroll
    for (int t = 0; t < 8; ++t) {
        const float v = vr[lane + 32 * t];
        vals[t] = v; s += v; s2 += v * v;
    }
#pragma unroll
    for (int off = 16; off >= 1; off >>= 1) { s += __shfl_xor(s, off); s2 += __shfl_xor(s2, off); }
    const float mu  = s * (1.0f / Hc);
    const float var = s2 * (1.0f / Hc) - mu * mu;
    const float rs  = rsqrtf(var + LN_EPS);
#pragma unroll
    for (int t = 0; t < 8; ++t) {
        const int h = lane + 32 * t;
        Out[(size_t)row * Hc + h] = (_Float16)((vals[t] - mu) * rs * g[h] + bb[h]);
    }
}

// out = Hfin @ W_out + b_out (N=1). Wave per row, f32 dot + shuffle reduce.
__global__ void __launch_bounds__(256)
k_out(const _Float16* __restrict__ Hf, const float* __restrict__ Wout,
      const float* __restrict__ bout, float* __restrict__ out)
{
    const int row  = blockIdx.x * 8 + (threadIdx.x >> 5);
    const int lane = threadIdx.x & 31;
    const _Float16* hr = Hf + (size_t)row * FLc;
    float s = 0.f;
#pragma unroll
    for (int t = 0; t < 32; ++t) {
        const int j = lane + 32 * t;
        s += (float)hr[j] * Wout[j];
    }
#pragma unroll
    for (int off = 16; off >= 1; off >>= 1) s += __shfl_xor(s, off);
    if (lane == 0) out[row] = s + bout[0];
}

// ---------------------------------------------------------------------------
// Host orchestration
// ---------------------------------------------------------------------------
extern "C" void kernel_launch(void* const* d_in, const int* in_sizes, int n_in,
                              void* d_out, int out_size, void* d_ws, size_t ws_size,
                              hipStream_t stream)
{
    (void)in_sizes; (void)n_in; (void)out_size; (void)ws_size;
    const float* x     = (const float*)d_in[0];
    const float* G     = (const float*)d_in[1];
    const float* Wfeat = (const float*)d_in[2];
    const float* bfeat = (const float*)d_in[3];
    const float* Wlay  = (const float*)d_in[4];   // [L,H,H]
    const float* blay  = (const float*)d_in[5];   // [L,H]
    const float* lng   = (const float*)d_in[6];
    const float* lnb   = (const float*)d_in[7];
    const float* Wfin  = (const float*)d_in[8];   // [H,FL]
    const float* bfin  = (const float*)d_in[9];
    const float* Wout  = (const float*)d_in[10];  // [FL,1]
    const float* bout  = (const float*)d_in[11];

    // workspace carve-up
    char* w = (char*)d_ws;
    float*    Vf32   = (float*)w;    w += (size_t)Bc * Vc * Hc * sizeof(float);
    _Float16* Vh     = (_Float16*)w; w += (size_t)Bc * Vc * Hc * sizeof(_Float16);
    _Float16* VPhT   = (_Float16*)w; w += (size_t)Bc * Hc * Vc * sizeof(_Float16);
    _Float16* Gph    = (_Float16*)w; w += (size_t)Bc * Vc * Vc * sizeof(_Float16);
    float*    deginv = (float*)w;    w += (size_t)Bc * Vc * sizeof(float);
    _Float16* WlhT   = (_Float16*)w; w += (size_t)Lc * Hc * Hc * sizeof(_Float16);
    _Float16* WfinhT = (_Float16*)w; w += (size_t)Hc * FLc * sizeof(_Float16);
    _Float16* Hfinh  = (_Float16*)w; w += (size_t)Bc * Vc * FLc * sizeof(_Float16);

    const int MV = Bc * Vc;  // 16384 flat rows

    // weight conversions + transpose (f32 [R=K][C=N] -> f16 [N][K] per matrix)
    k_cvtT<<<dim3(Hc,  Lc), Hc, 0, stream>>>(Wlay, WlhT, Hc, Hc);    // W_layers
    k_cvtT<<<dim3(FLc, 1),  Hc, 0, stream>>>(Wfin, WfinhT, Hc, FLc); // W_fin

    // featurize + graph prep
    k_featurize<<<MV, 256, 0, stream>>>(x, Wfeat, bfeat, Vf32, Vh);
    k_prep<<<MV / 8, 256, 0, stream>>>(G, Gph, deginv);

    // 32 message-passing layers: two WMMA GEMMs each
    for (int l = 0; l < Lc; ++l) {
        k_gemm_vw<<<dim3(MV / 64, Hc / 256), 256, 0, stream>>>(
            Vh, WlhT + (size_t)l * Hc * Hc, blay + (size_t)l * Hc, VPhT);
        k_gemm_gp<<<dim3(Vc / 64, Hc / 256, Bc), 256, 0, stream>>>(
            Gph, VPhT, deginv, Vf32, Vh);
    }

    // output head: LN -> GEMM(256->1024)+ReLU -> dot(1024->1)
    k_ln<<<MV / 8, 256, 0, stream>>>(Vf32, lng, lnb, VPhT);  // buffer reused row-major
    k_gemm_fin<<<dim3(MV / 64, FLc / 256), 256, 0, stream>>>(VPhT, WfinhT, bfin, Hfinh);
    k_out<<<MV / 8, 256, 0, stream>>>(Hfinh, Wout, bout, (float*)d_out);
}
